// VoxelWithPointProjectionV2KITTI_68839735820803
// MI455X (gfx1250) — compile-verified, matrix operand order checked by reading, not run on hardware
//
#include <hip/hip_runtime.h>
#include <cstdint>

// Problem constants
#define N_PTS 120000
#define C 128
#define NH 4
#define NP 4
#define DH 32
#define H_IMG 192
#define W_IMG 640
#define HW (H_IMG * W_IMG)
#define DFF 512

typedef __attribute__((ext_vector_type(16))) _Float16 v16h;
typedef __attribute__((ext_vector_type(8)))  float    v8f;
typedef _Float16 half_t;

// ---------------------------------------------------------------------------
// WMMA helpers (CDNA5 wave32, 16x16x32 f16 -> f32)
// A-fragment (16xK tile) & B-fragment (Kx16 tile of row-major W[Nout][K]) have
// the same per-lane indexing: lane L handles row/col (L&15); lanes 0-15 hold
// K = k0+0..7 then k0+16..23, lanes 16-31 hold K = k0+8..15 then k0+24..31.
// ---------------------------------------------------------------------------
__device__ inline v8f wmma16(v16h a, v16h b, v8f c) {
  return __builtin_amdgcn_wmma_f32_16x16x32_f16(
      /*neg_a=*/false, a, /*neg_b=*/false, b,
      /*c_mod=*/(short)0, c, /*reuse_a=*/false, /*reuse_b=*/false);
}

// fragment from row-major f32 matrix (converted to f16 on load)
__device__ inline v16h frag_f32(const float* __restrict__ M, int ld, int r0, int k0) {
  const int lane = threadIdx.x & 31;
  const int hs = lane >> 4;
  const float* p = M + (size_t)(r0 + (lane & 15)) * ld + k0 + hs * 8;
  v16h f;
#pragma unroll
  for (int i = 0; i < 8; ++i) f[i] = (_Float16)p[i];
#pragma unroll
  for (int i = 0; i < 8; ++i) f[8 + i] = (_Float16)p[16 + i];
  return f;
}

// fragment from row-major f16 matrix
__device__ inline v16h frag_h16(const half_t* __restrict__ M, int ld, int r0, int k0) {
  const int lane = threadIdx.x & 31;
  const int hs = lane >> 4;
  const half_t* p = M + (size_t)(r0 + (lane & 15)) * ld + k0 + hs * 8;
  v16h f;
#pragma unroll
  for (int i = 0; i < 8; ++i) f[i] = p[i];
#pragma unroll
  for (int i = 0; i < 8; ++i) f[8 + i] = p[16 + i];
  return f;
}

// A-fragment where A[m][k] = IF[k*HW + p0+m]  (image_feat columns)
__device__ inline v16h frag_imgfeat(const float* __restrict__ IF, int p0, int k0) {
  const int lane = threadIdx.x & 31;
  const int hs = lane >> 4;
  const int p = p0 + (lane & 15);
  const int kb = k0 + hs * 8;
  v16h f;
#pragma unroll
  for (int i = 0; i < 8; ++i) f[i] = (_Float16)IF[(size_t)(kb + i) * HW + p];
#pragma unroll
  for (int i = 0; i < 8; ++i) f[8 + i] = (_Float16)IF[(size_t)(kb + 16 + i) * HW + p];
  return f;
}

// ---------------------------------------------------------------------------
// K0: f32 -> f16 weight conversion
// ---------------------------------------------------------------------------
__global__ void cvt_f16_kernel(const float* __restrict__ src, half_t* __restrict__ dst, int n) {
  int i = blockIdx.x * blockDim.x + threadIdx.x;
  if (i < n) dst[i] = (half_t)src[i];
}

// ---------------------------------------------------------------------------
// K1: grid statistics (center + max |rel|) via exact integer atomics
// ---------------------------------------------------------------------------
__global__ void stats_init_kernel(int* st) {
  if (threadIdx.x == 0) {
    st[0] = 0; st[1] = 0;                     // sums
    st[2] = 0x7fffffff; st[3] = 0x7fffffff;   // mins
    st[4] = (int)0x80000000; st[5] = (int)0x80000000; // maxs
  }
}

__global__ void stats_reduce_kernel(const int* __restrict__ grid, int* st) {
  int i = blockIdx.x * blockDim.x + threadIdx.x;
  if (i >= N_PTS) return;
  int gx = grid[2 * i], gy = grid[2 * i + 1];
  atomicAdd(&st[0], gx); atomicAdd(&st[1], gy);
  atomicMin(&st[2], gx); atomicMin(&st[3], gy);
  atomicMax(&st[4], gx); atomicMax(&st[5], gy);
}

__global__ void stats_final_kernel(const int* __restrict__ st, float* out4) {
  if (threadIdx.x == 0) {
    float cx = (float)st[0] / (float)N_PTS;
    float cy = (float)st[1] / (float)N_PTS;
    float mxx = fmaxf(fmaxf(cx - (float)st[2], (float)st[4] - cx), 1.0f);
    float mxy = fmaxf(fmaxf(cy - (float)st[3], (float)st[5] - cy), 1.0f);
    out4[0] = cx; out4[1] = cy; out4[2] = mxx; out4[3] = mxy;
  }
}

// ---------------------------------------------------------------------------
// K2: vmap[p][c] = sum_k IF[k][p] * val_w[c][k] + val_b[c]   (M=HW, K=C, N=C)
// 32 rows per wave: one B-fragment feeds two WMMAs.
// ---------------------------------------------------------------------------
__global__ __launch_bounds__(32)
void vmap_kernel(const float* __restrict__ image_feat,
                 const half_t* __restrict__ valw16,
                 const float* __restrict__ val_b,
                 float* __restrict__ vmap) {
  const int p0 = blockIdx.x * 32;
  const int lane = threadIdx.x;
  const int col = lane & 15, hs = lane >> 4;
  v16h a0[4], a1[4];
#pragma unroll
  for (int ks = 0; ks < 4; ++ks) {
    a0[ks] = frag_imgfeat(image_feat, p0, ks * 32);
    a1[ks] = frag_imgfeat(image_feat, p0 + 16, ks * 32);
  }
  for (int nt = 0; nt < 8; ++nt) {
    v8f acc0 = {}, acc1 = {};
#pragma unroll
    for (int ks = 0; ks < 4; ++ks) {
      const v16h b = frag_h16(valw16, C, nt * 16, ks * 32);
      acc0 = wmma16(a0[ks], b, acc0);
      acc1 = wmma16(a1[ks], b, acc1);
    }
    const int n = nt * 16 + col;
    const float b = val_b[n];
#pragma unroll
    for (int r = 0; r < 8; ++r) {
      const int m = r + 8 * hs;
      vmap[(size_t)(p0 + m) * C + n] = acc0[r] + b;
      vmap[(size_t)(p0 + 16 + m) * C + n] = acc1[r] + b;
    }
  }
}

// ---------------------------------------------------------------------------
// K3: query = affine(pf@pkp_w^T) + affine(rel@fc_w^T) + image_feat gather
// 32 rows per wave.
// ---------------------------------------------------------------------------
__global__ __launch_bounds__(32)
void query_kernel(const float* __restrict__ pf, const half_t* __restrict__ pkpw16,
                  const float* __restrict__ pkp_b, const float* __restrict__ pkp_s,
                  const float* __restrict__ pkp_t,
                  const float* __restrict__ fc_w, const float* __restrict__ fc_b,
                  const float* __restrict__ fc_s, const float* __restrict__ fc_t,
                  const float* __restrict__ image_feat, const int* __restrict__ grid,
                  const float* __restrict__ stats,
                  float* __restrict__ query) {
  const int m0 = blockIdx.x * 32;
  const int lane = threadIdx.x;
  const int col = lane & 15, hs = lane >> 4;
  v16h a0[4], a1[4];
#pragma unroll
  for (int ks = 0; ks < 4; ++ks) {
    a0[ks] = frag_f32(pf, C, m0, ks * 32);
    a1[ks] = frag_f32(pf, C, m0 + 16, ks * 32);
  }
  const float cx = stats[0], cy = stats[1], mxx = stats[2], mxy = stats[3];
  for (int nt = 0; nt < 8; ++nt) {
    v8f acc0 = {}, acc1 = {};
#pragma unroll
    for (int ks = 0; ks < 4; ++ks) {
      const v16h b = frag_h16(pkpw16, C, nt * 16, ks * 32);
      acc0 = wmma16(a0[ks], b, acc0);
      acc1 = wmma16(a1[ks], b, acc1);
    }
    const int n = nt * 16 + col;
    const float bb = pkp_b[n], s = pkp_s[n], t = pkp_t[n];
    const float w0 = fc_w[n * 2 + 0], w1 = fc_w[n * 2 + 1];
    const float fb = fc_b[n], fs = fc_s[n], ft = fc_t[n];
#pragma unroll
    for (int rt = 0; rt < 2; ++rt) {
#pragma unroll
      for (int r = 0; r < 8; ++r) {
        const int m = m0 + rt * 16 + r + 8 * hs;
        const int gx = grid[m * 2 + 0], gy = grid[m * 2 + 1];
        const float rx = ((float)gx - cx) / mxx;
        const float ry = ((float)gy - cy) / mxy;
        const float pos = (rx * w0 + ry * w1 + fb) * fs + ft;
        const float acc = (rt == 0) ? acc0[r] : acc1[r];
        const float pts = (acc + bb) * s + t;
        const float ff = image_feat[(size_t)n * HW + gy * W_IMG + gx];
        query[(size_t)m * C + n] = pts + pos + ff;
      }
    }
  }
}

// ---------------------------------------------------------------------------
// K4: deformable attention + residual + LayerNorm1 (16 rows per wave)
// ---------------------------------------------------------------------------
__global__ __launch_bounds__(32)
void attn_kernel(const float* __restrict__ query, const int* __restrict__ grid,
                 const half_t* __restrict__ offw16, const float* __restrict__ off_b,
                 const half_t* __restrict__ aww16, const float* __restrict__ aw_b,
                 const float* __restrict__ vmap,
                 const half_t* __restrict__ outw16, const float* __restrict__ out_b,
                 const float* __restrict__ ln1_g, const float* __restrict__ ln1_b,
                 float* __restrict__ qout) {
  __shared__ __align__(16) float s_off[16][NH * NP * 2];
  __shared__ __align__(16) float s_aw[16][NH * NP];
  __shared__ __align__(16) float s_samp[16][C];
  __shared__ __align__(16) float s_res[16][C];
  const int m0 = blockIdx.x * 16;
  const int lane = threadIdx.x;
  const int col = lane & 15, hs = lane >> 4;

  v16h a[4];
#pragma unroll
  for (int ks = 0; ks < 4; ++ks) a[ks] = frag_f32(query, C, m0, ks * 32);

  // offsets: 2 column tiles of 16
#pragma unroll
  for (int nt = 0; nt < 2; ++nt) {
    v8f acc = {};
#pragma unroll
    for (int ks = 0; ks < 4; ++ks)
      acc = wmma16(a[ks], frag_h16(offw16, C, nt * 16, ks * 32), acc);
    const int n = nt * 16 + col;
    const float b = off_b[n];
#pragma unroll
    for (int r = 0; r < 8; ++r) s_off[r + 8 * hs][n] = acc[r] + b;
  }
  // attention logits: 1 column tile of 16
  {
    v8f acc = {};
#pragma unroll
    for (int ks = 0; ks < 4; ++ks)
      acc = wmma16(a[ks], frag_h16(aww16, C, 0, ks * 32), acc);
    const float b = aw_b[col];
#pragma unroll
    for (int r = 0; r < 8; ++r) s_aw[r + 8 * hs][col] = acc[r] + b;
  }
  __syncthreads();

  // bilinear sampling: 64 (row, head) combos over 32 lanes.
  // px = locs.x*W - 0.5 = gx + off_x exactly (the 0.5's cancel); same for py.
  for (int t = lane; t < 64; t += 32) {
    const int m = t & 15, h = t >> 4;
    const int gx = grid[(m0 + m) * 2 + 0], gy = grid[(m0 + m) * 2 + 1];
    const float l0 = s_aw[m][h * 4 + 0], l1 = s_aw[m][h * 4 + 1];
    const float l2 = s_aw[m][h * 4 + 2], l3 = s_aw[m][h * 4 + 3];
    const float lm = fmaxf(fmaxf(l0, l1), fmaxf(l2, l3));
    const float e0 = __expf(l0 - lm), e1 = __expf(l1 - lm);
    const float e2 = __expf(l2 - lm), e3 = __expf(l3 - lm);
    const float inv = 1.0f / (e0 + e1 + e2 + e3);
    const float wsm[4] = {e0 * inv, e1 * inv, e2 * inv, e3 * inv};

    // compute all 16 corner pointers + weights, prefetch the lines,
    // then accumulate. Invalid corners: weight 0 + clamped pointer (exact,
    // since vmap is finite).
    const float* vp[16];
    float wgt[16];
#pragma unroll
    for (int p = 0; p < NP; ++p) {
      const float px = (float)gx + s_off[m][(h * 4 + p) * 2 + 0];
      const float py = (float)gy + s_off[m][(h * 4 + p) * 2 + 1];
      const float x0f = floorf(px), y0f = floorf(py);
      const float dx = px - x0f, dy = py - y0f;
      const int x0 = (int)x0f, y0 = (int)y0f;
      const float cw[4] = {(1.f - dx) * (1.f - dy), dx * (1.f - dy),
                           (1.f - dx) * dy, dx * dy};
      const int xs[4] = {x0, x0 + 1, x0, x0 + 1};
      const int ys[4] = {y0, y0, y0 + 1, y0 + 1};
#pragma unroll
      for (int cI = 0; cI < 4; ++cI) {
        const int xi = xs[cI], yi = ys[cI];
        const bool valid = (xi >= 0) && (xi < W_IMG) && (yi >= 0) && (yi < H_IMG);
        const int xc = min(max(xi, 0), W_IMG - 1);
        const int yc = min(max(yi, 0), H_IMG - 1);
        const int idx = p * 4 + cI;
        vp[idx] = vmap + ((size_t)(yc * W_IMG + xc) * C + h * DH);
        wgt[idx] = valid ? (wsm[p] * cw[cI]) : 0.0f;
        __builtin_prefetch(vp[idx], 0, 3);  // global_prefetch_b8
      }
    }
    float acc[DH];
#pragma unroll
    for (int d = 0; d < DH; ++d) acc[d] = 0.0f;
#pragma unroll
    for (int idx = 0; idx < 16; ++idx) {
      const float w = wgt[idx];
      const float* v = vp[idx];
#pragma unroll
      for (int d = 0; d < DH; ++d) acc[d] += w * v[d];
    }
#pragma unroll
    for (int d = 0; d < DH; ++d) s_samp[m][h * DH + d] = acc[d];
  }
  __syncthreads();

  // attn = sampled @ out_w^T + out_b ; residual with query -> s_res
  v16h sa[4];
#pragma unroll
  for (int ks = 0; ks < 4; ++ks) sa[ks] = frag_f32(&s_samp[0][0], C, 0, ks * 32);
  for (int nt = 0; nt < 8; ++nt) {
    v8f acc = {};
#pragma unroll
    for (int ks = 0; ks < 4; ++ks)
      acc = wmma16(sa[ks], frag_h16(outw16, C, nt * 16, ks * 32), acc);
    const int n = nt * 16 + col;
    const float b = out_b[n];
#pragma unroll
    for (int r = 0; r < 8; ++r) {
      const int m = r + 8 * hs;
      s_res[m][n] = acc[r] + b + query[(size_t)(m0 + m) * C + n];
    }
  }
  __syncthreads();

  // LayerNorm1 (one lane per row, lanes 0..15)
  if (lane < 16) {
    const int m = lane;
    float mean = 0.f;
    for (int c = 0; c < C; ++c) mean += s_res[m][c];
    mean *= (1.0f / C);
    float var = 0.f;
    for (int c = 0; c < C; ++c) { float d = s_res[m][c] - mean; var += d * d; }
    var *= (1.0f / C);
    const float inv = rsqrtf(var + 1e-5f);
    for (int c = 0; c < C; ++c)
      qout[(size_t)(m0 + m) * C + c] = (s_res[m][c] - mean) * inv * ln1_g[c] + ln1_b[c];
  }
}

// ---------------------------------------------------------------------------
// K5: FFN (128 -> 512 relu -> 128) + residual + LayerNorm2 (32 rows per wave)
// ---------------------------------------------------------------------------
__global__ __launch_bounds__(32)
void ffn_kernel(const float* __restrict__ q, const half_t* __restrict__ w1_16,
                const float* __restrict__ b1, const half_t* __restrict__ w2_16,
                const float* __restrict__ b2,
                const float* __restrict__ ln2_g, const float* __restrict__ ln2_b,
                float* __restrict__ qout) {
  __shared__ __align__(16) half_t s_hid[32][DFF];   // 32 KB
  __shared__ __align__(16) float s_res[32][C];      // 16 KB
  const int m0 = blockIdx.x * 32;
  const int lane = threadIdx.x;
  const int col = lane & 15, hs = lane >> 4;

  v16h a0[4], a1[4];
#pragma unroll
  for (int ks = 0; ks < 4; ++ks) {
    a0[ks] = frag_f32(q, C, m0, ks * 32);
    a1[ks] = frag_f32(q, C, m0 + 16, ks * 32);
  }

  // hidden = relu(q @ w1^T + b1), stored to LDS as f16 (A for next GEMM)
  for (int nt = 0; nt < DFF / 16; ++nt) {
    v8f acc0 = {}, acc1 = {};
#pragma unroll
    for (int ks = 0; ks < 4; ++ks) {
      const v16h b = frag_h16(w1_16, C, nt * 16, ks * 32);
      acc0 = wmma16(a0[ks], b, acc0);
      acc1 = wmma16(a1[ks], b, acc1);
    }
    const int n = nt * 16 + col;
    const float b = b1[n];
#pragma unroll
    for (int r = 0; r < 8; ++r) {
      const int m = r + 8 * hs;
      s_hid[m][n] = (half_t)fmaxf(acc0[r] + b, 0.0f);
      s_hid[16 + m][n] = (half_t)fmaxf(acc1[r] + b, 0.0f);
    }
  }
  __syncthreads();

  // out = hidden @ w2^T + b2 ; residual with q
  for (int nt = 0; nt < 8; ++nt) {
    v8f acc0 = {}, acc1 = {};
    for (int ks = 0; ks < DFF / 32; ++ks) {
      const v16h b = frag_h16(w2_16, DFF, nt * 16, ks * 32);
      acc0 = wmma16(frag_h16(&s_hid[0][0], DFF, 0, ks * 32), b, acc0);
      acc1 = wmma16(frag_h16(&s_hid[0][0], DFF, 16, ks * 32), b, acc1);
    }
    const int n = nt * 16 + col;
    const float b = b2[n];
#pragma unroll
    for (int r = 0; r < 8; ++r) {
      const int m = r + 8 * hs;
      s_res[m][n] = acc0[r] + b + q[(size_t)(m0 + m) * C + n];
      s_res[16 + m][n] = acc1[r] + b + q[(size_t)(m0 + 16 + m) * C + n];
    }
  }
  __syncthreads();

  // LayerNorm2: one lane per row, all 32 lanes busy
  {
    const int m = lane;
    float mean = 0.f;
    for (int c = 0; c < C; ++c) mean += s_res[m][c];
    mean *= (1.0f / C);
    float var = 0.f;
    for (int c = 0; c < C; ++c) { float d = s_res[m][c] - mean; var += d * d; }
    var *= (1.0f / C);
    const float inv = rsqrtf(var + 1e-5f);
    for (int c = 0; c < C; ++c)
      qout[(size_t)(m0 + m) * C + c] = (s_res[m][c] - mean) * inv * ln2_g[c] + ln2_b[c];
  }
}

// ---------------------------------------------------------------------------
// K6: pts_pre GEMM, relu-concat, fuse GEMM (K=256), affine + relu -> out
// 32 rows per wave.
// ---------------------------------------------------------------------------
__global__ __launch_bounds__(32)
void final_kernel(const float* __restrict__ pf, const half_t* __restrict__ ptw16,
                  const float* __restrict__ pt_b, const float* __restrict__ pt_s,
                  const float* __restrict__ pt_t,
                  const float* __restrict__ q2, const half_t* __restrict__ fusew16,
                  const float* __restrict__ fuse_b, const float* __restrict__ fuse_s,
                  const float* __restrict__ fuse_t,
                  float* __restrict__ out) {
  __shared__ __align__(16) half_t s_fused[32][2 * C];  // 16 KB
  const int m0 = blockIdx.x * 32;
  const int lane = threadIdx.x;
  const int col = lane & 15, hs = lane >> 4;

  v16h a0[4], a1[4];
#pragma unroll
  for (int ks = 0; ks < 4; ++ks) {
    a0[ks] = frag_f32(pf, C, m0, ks * 32);
    a1[ks] = frag_f32(pf, C, m0 + 16, ks * 32);
  }

  // relu(affine(pf @ pt_w^T)) -> s_fused[:, 0:128]
  for (int nt = 0; nt < 8; ++nt) {
    v8f acc0 = {}, acc1 = {};
#pragma unroll
    for (int ks = 0; ks < 4; ++ks) {
      const v16h b = frag_h16(ptw16, C, nt * 16, ks * 32);
      acc0 = wmma16(a0[ks], b, acc0);
      acc1 = wmma16(a1[ks], b, acc1);
    }
    const int n = nt * 16 + col;
    const float b = pt_b[n], s = pt_s[n], t = pt_t[n];
#pragma unroll
    for (int r = 0; r < 8; ++r) {
      const int m = r + 8 * hs;
      s_fused[m][n] = (half_t)fmaxf((acc0[r] + b) * s + t, 0.0f);
      s_fused[16 + m][n] = (half_t)fmaxf((acc1[r] + b) * s + t, 0.0f);
    }
  }
  // relu(q2) -> s_fused[:, 128:256]
  for (int i = lane; i < 32 * C; i += 32) {
    const int m = i >> 7, c = i & (C - 1);
    s_fused[m][C + c] = (half_t)fmaxf(q2[(size_t)(m0 + m) * C + c], 0.0f);
  }
  __syncthreads();

  // out = relu(affine(fused @ fuse_w^T + fuse_b))
  for (int nt = 0; nt < 8; ++nt) {
    v8f acc0 = {}, acc1 = {};
    for (int ks = 0; ks < (2 * C) / 32; ++ks) {
      const v16h b = frag_h16(fusew16, 2 * C, nt * 16, ks * 32);
      acc0 = wmma16(frag_h16(&s_fused[0][0], 2 * C, 0, ks * 32), b, acc0);
      acc1 = wmma16(frag_h16(&s_fused[0][0], 2 * C, 16, ks * 32), b, acc1);
    }
    const int n = nt * 16 + col;
    const float b = fuse_b[n], s = fuse_s[n], t = fuse_t[n];
#pragma unroll
    for (int r = 0; r < 8; ++r) {
      const int m = r + 8 * hs;
      out[(size_t)(m0 + m) * C + n] = fmaxf((acc0[r] + b) * s + t, 0.0f);
      out[(size_t)(m0 + 16 + m) * C + n] = fmaxf((acc1[r] + b) * s + t, 0.0f);
    }
  }
}

// ---------------------------------------------------------------------------
// Launch
// ---------------------------------------------------------------------------
extern "C" void kernel_launch(void* const* d_in, const int* in_sizes, int n_in,
                              void* d_out, int out_size, void* d_ws, size_t ws_size,
                              hipStream_t stream) {
  const float* pf       = (const float*)d_in[0];
  const float* img_feat = (const float*)d_in[1];
  const int*   grid     = (const int*)d_in[2];
  const float* pkp_w = (const float*)d_in[3];
  const float* pkp_b = (const float*)d_in[4];
  const float* pkp_s = (const float*)d_in[5];
  const float* pkp_t = (const float*)d_in[6];
  const float* fc_w  = (const float*)d_in[7];
  const float* fc_b  = (const float*)d_in[8];
  const float* fc_s  = (const float*)d_in[9];
  const float* fc_t  = (const float*)d_in[10];
  const float* pt_w  = (const float*)d_in[11];
  const float* pt_b  = (const float*)d_in[12];
  const float* pt_s  = (const float*)d_in[13];
  const float* pt_t  = (const float*)d_in[14];
  const float* val_w = (const float*)d_in[15];
  const float* val_b = (const float*)d_in[16];
  const float* off_w = (const float*)d_in[17];
  const float* off_b = (const float*)d_in[18];
  const float* aw_w  = (const float*)d_in[19];
  const float* aw_b  = (const float*)d_in[20];
  const float* out_w = (const float*)d_in[21];
  const float* out_b = (const float*)d_in[22];
  const float* ln1_g = (const float*)d_in[23];
  const float* ln1_b = (const float*)d_in[24];
  const float* ffn_w1 = (const float*)d_in[25];
  const float* ffn_b1 = (const float*)d_in[26];
  const float* ffn_w2 = (const float*)d_in[27];
  const float* ffn_b2 = (const float*)d_in[28];
  const float* ln2_g = (const float*)d_in[29];
  const float* ln2_b = (const float*)d_in[30];
  const float* fuse_w = (const float*)d_in[31];
  const float* fuse_b = (const float*)d_in[32];
  const float* fuse_s = (const float*)d_in[33];
  const float* fuse_t = (const float*)d_in[34];

  // Workspace layout (needs ~187 MB):
  //  [0, 471040)              f16 weights
  //  [471040, 471104)         int stats (6 ints)
  //  [471104, 471168)         float stats (4 floats) + pad
  //  [471168, ...)            float buffers: query | vmap | q_ln1
  char* ws = (char*)d_ws;
  half_t* pkpw16  = (half_t*)ws;
  half_t* ptw16   = pkpw16 + C * C;        // 16384
  half_t* valw16  = ptw16  + C * C;
  half_t* offw16  = valw16 + C * C;
  half_t* aww16   = offw16 + (NH * NP * 2) * C;  // 4096
  half_t* outw16  = aww16  + (NH * NP) * C;      // 2048
  half_t* w1_16   = outw16 + C * C;
  half_t* w2_16   = w1_16  + DFF * C;            // 65536
  half_t* fusew16 = w2_16  + C * DFF;            // 65536
  // fusew16 spans 32768 halves -> total 235520 halves = 471040 bytes
  int*   stats_i = (int*)(ws + 471040);
  float* stats_f = (float*)(ws + 471104);
  float* queryB  = (float*)(ws + 471168);
  float* vmapB   = queryB + (size_t)N_PTS * C;
  float* qB      = vmapB + (size_t)HW * C;
  float* outF    = (float*)d_out;

  // K0: weight conversion
  auto cvt = [&](const float* s, half_t* d, int n) {
    cvt_f16_kernel<<<(n + 255) / 256, 256, 0, stream>>>(s, d, n);
  };
  cvt(pkp_w, pkpw16, C * C);
  cvt(pt_w,  ptw16,  C * C);
  cvt(val_w, valw16, C * C);
  cvt(off_w, offw16, (NH * NP * 2) * C);
  cvt(aw_w,  aww16,  (NH * NP) * C);
  cvt(out_w, outw16, C * C);
  cvt(ffn_w1, w1_16, DFF * C);
  cvt(ffn_w2, w2_16, C * DFF);
  cvt(fuse_w, fusew16, C * 2 * C);

  // K1: grid stats
  stats_init_kernel<<<1, 32, 0, stream>>>(stats_i);
  stats_reduce_kernel<<<(N_PTS + 255) / 256, 256, 0, stream>>>(grid, stats_i);
  stats_final_kernel<<<1, 32, 0, stream>>>(stats_i, stats_f);

  // K2: value map (122880 x 128), 32 rows per block
  vmap_kernel<<<HW / 32, 32, 0, stream>>>(img_feat, valw16, val_b, vmapB);

  // K3: query, 32 rows per block
  query_kernel<<<N_PTS / 32, 32, 0, stream>>>(
      pf, pkpw16, pkp_b, pkp_s, pkp_t, fc_w, fc_b, fc_s, fc_t,
      img_feat, grid, stats_f, queryB);

  // K4: deformable attention + LN1, 16 rows per block
  attn_kernel<<<N_PTS / 16, 32, 0, stream>>>(
      queryB, grid, offw16, off_b, aww16, aw_b, vmapB,
      outw16, out_b, ln1_g, ln1_b, qB);

  // K5: FFN + LN2 (writes q2 into queryB, which is dead after K4)
  ffn_kernel<<<N_PTS / 32, 32, 0, stream>>>(
      qB, w1_16, ffn_b1, w2_16, ffn_b2, ln2_g, ln2_b, queryB);

  // K6: pt branch + fuse -> d_out, 32 rows per block
  final_kernel<<<N_PTS / 32, 32, 0, stream>>>(
      pf, ptw16, pt_b, pt_s, pt_t, queryB, fusew16,
      fuse_b, fuse_s, fuse_t, outF);
}
